// ImitationBlock_35510789603985
// MI455X (gfx1250) — matrix operand order repaired and checked by reading, not compiled, not run on hardware
//
#include <hip/hip_runtime.h>
#include <hip/hip_bf16.h>

// ---------------- problem constants ----------------
#define NTHREADS 256
#define NWAVES   8
#define MT       64      // rows per workgroup tile
#define RTILES   4       // 16-row WMMA tiles covering MT

#define S_DIM  256
#define G_DIM  128
#define L_DIM  128
#define H_DIM  512
#define DH_DIM 1024
#define A_DIM  32
#define QN     8
#define KCB    1024
#define CHUNK_N 256

typedef __attribute__((ext_vector_type(16))) __bf16 v16bf;
typedef __attribute__((ext_vector_type(8)))  __bf16 v8bf;
typedef __attribute__((ext_vector_type(8)))  float  v8f;

// ---------------- LDS layout (bytes) ----------------
#define OFF_SBF 0
#define SZ_SBF  (MT*S_DIM*2)            // 32768  s tile, bf16
#define OFF_GBF (OFF_SBF + SZ_SBF)
#define SZ_GBF  (MT*G_DIM*2)            // 16384  goal tile, bf16
#define OFF_H0  (OFF_GBF + SZ_GBF)
#define SZ_H0   (MT*H_DIM*2)            // 65536  hidden ping, bf16
#define OFF_H1  (OFF_H0 + SZ_H0)
#define SZ_H1   (MT*H_DIM*2)            // 65536  hidden pong (H0+H1 -> 64x1024 decoder hidden)
#define OFF_ZP  (OFF_H1 + SZ_H1)
#define SZ_ZP   (MT*L_DIM*4)            // 32768  zp -> z_hat accumulator, f32
#define OFF_Z   (OFF_ZP + SZ_ZP)
#define SZ_Z    (MT*L_DIM*4)            // 32768  z  -> residual, f32
#define OFF_MIN (OFF_Z + SZ_Z)
#define SZ_MIN  (MT*8)                  // 512    per-row packed argmin (u64)
#define SMEM_TOTAL (OFF_MIN + SZ_MIN)   // 246272 bytes (< 320KB WGP LDS)
// region reuse (sequential lifetimes):
//   RBF  (64x128 bf16, residual / z_hat mirror)  overlays GBF  (goal dead after posterior L0)
//   CHUNK(64x256 bf16)                           overlays SBF  (s dead after decoder L0)
//   AACC (64x32 f32)                             overlays GBF  (after decoder L0)

// ---------------- packed-weight workspace layout (bf16 elements) ----------------
// W[K,C] fp32  ->  WT[C,K] bf16  (B-fragment becomes 2 contiguous b128 loads per lane)
#define OFS_PW0T 0
#define OFS_PW1T (OFS_PW0T + 512*256)
#define OFS_PW2T (OFS_PW1T + 512*512)
#define OFS_QW0T (OFS_PW2T + 128*512)
#define OFS_QW1T (OFS_QW0T + 512*384)
#define OFS_QW2T (OFS_QW1T + 512*512)
#define OFS_DW0T (OFS_QW2T + 128*512)
#define OFS_DW1T (OFS_DW0T + 1024*384)
#define OFS_DW2T (OFS_DW1T + 1024*1024)
#define OFS_CBBF (OFS_DW2T + 32*1024)
#define OFS_END  (OFS_CBBF + QN*KCB*L_DIM)
#define OFS_CB2B ((size_t)OFS_END * 2)              // byte offset of fp32 ||cb||^2 table
#define WS_NEED  (OFS_CB2B + (size_t)QN*KCB*4)      // 7,045,120 bytes

// ---------------- WMMA fragment loaders ----------------
// A: 16x32 bf16. lanes 0-15: row=lane, K {k0+0..15}; lanes 16-31: row=lane-16, K {k0+16..31}
static __device__ __forceinline__ v16bf frag_a_bf(const __bf16* X, int ld, int row0, int k0, int lane) {
  int r  = row0 + (lane & 15);
  int kk = k0 + ((lane >> 4) << 4);
  const __bf16* p = X + r * ld + kk;
  v8bf lo = *(const v8bf*)p;
  v8bf hi = *(const v8bf*)(p + 8);
  v16bf a;
#pragma unroll
  for (int i = 0; i < 8; ++i) { a[i] = lo[i]; a[i + 8] = hi[i]; }
  return a;
}

// B (packed): WT[C,K] bf16, lane = column, 16 contiguous K values -> 2x b128
static __device__ __forceinline__ v16bf frag_b_bf(const __bf16* WT, int ld, int k0, int n0, int lane) {
  int c  = n0 + (lane & 15);
  int kk = k0 + ((lane >> 4) << 4);
  const __bf16* p = WT + (size_t)c * ld + kk;
  v8bf lo = *(const v8bf*)p;
  v8bf hi = *(const v8bf*)(p + 8);
  v16bf b;
#pragma unroll
  for (int i = 0; i < 8; ++i) { b[i] = lo[i]; b[8 + i] = hi[i]; }
  return b;
}

// B (fallback): row-major fp32 W[K, ncols], strided scalar loads + cvt
static __device__ __forceinline__ v16bf frag_b_w(const float* W, int wld, int k0, int n0, int lane) {
  int c  = n0 + (lane & 15);
  int kk = k0 + ((lane >> 4) << 4);
  const float* p = W + (size_t)kk * wld + c;
  v16bf b;
#pragma unroll
  for (int i = 0; i < 16; ++i) b[i] = (__bf16)p[(size_t)i * wld];
  return b;
}

// B (fallback) for distance GEMM: B[k=l, n=code] = cb[code, l] (contiguous fp32 per code)
static __device__ __forceinline__ v16bf frag_b_cb(const float* cb, int k0, int c0, int lane) {
  int code = c0 + (lane & 15);
  int kk   = k0 + ((lane >> 4) << 4);
  const float* p = cb + (size_t)code * L_DIM + kk;
  v16bf b;
#pragma unroll
  for (int i = 0; i < 16; ++i) b[i] = (__bf16)p[i];
  return b;
}

// A-source with optional K-split (for concat inputs); both regions bf16 in LDS
struct ASrc { const __bf16* p0; int ld0; int ksplit; const __bf16* p1; int ld1; };

static __device__ __forceinline__ v16bf frag_a(const ASrc& S, int row0, int k0, int lane) {
  if (k0 < S.ksplit) return frag_a_bf(S.p0, S.ld0, row0, k0, lane);
  return frag_a_bf(S.p1, S.ld1, row0, k0 - S.ksplit, lane);
}

// B-source: packed bf16 [C,K] (t,ldt) or fp32 [K,C] (f,ldf); selected at compile time
struct BSrc { const __bf16* t; int ldt; const float* f; int ldf; };

template <bool PACKED>
static __device__ __forceinline__ v16bf frag_b(const BSrc& B, int k0, int n0, int lane) {
  if constexpr (PACKED) return frag_b_bf(B.t, B.ldt, k0, n0, lane);
  else                  return frag_b_w(B.f, B.ldf, k0, n0, lane);
}

// one dense layer: Y[MT, ncols] = act(A[MT,K] @ W[K,ncols] + bias)
// waves split the column tiles; each B fragment feeds RTILES WMMAs.
template <bool PACKED>
static __device__ __forceinline__ void dense_layer(
    const ASrc& A, int K, const BSrc& B, const float* bias,
    int ncols, __bf16* ybf, float* yf, bool relu, int wave, int lane)
{
  for (int ct = wave; ct < (ncols >> 4); ct += NWAVES) {
    int n0 = ct << 4;
    float bv = bias[n0 + (lane & 15)];
    v8f acc[RTILES];
#pragma unroll
    for (int rt = 0; rt < RTILES; ++rt)
#pragma unroll
      for (int j = 0; j < 8; ++j) acc[rt][j] = bv;

    for (int k0 = 0; k0 < K; k0 += 32) {
      v16bf b = frag_b<PACKED>(B, k0, n0, lane);
      if (k0 + 32 < K) {
        if constexpr (PACKED) __builtin_prefetch(B.t + (size_t)(n0 + (lane & 15)) * B.ldt + k0 + 32, 0, 1);
        else                  __builtin_prefetch(B.f + (size_t)(k0 + 32) * B.ldf + n0 + (lane & 15), 0, 1);
      }
#pragma unroll
      for (int rt = 0; rt < RTILES; ++rt) {
        v16bf a = frag_a(A, rt << 4, k0, lane);
        acc[rt] = __builtin_amdgcn_wmma_f32_16x16x32_bf16(false, a, false, b, (short)0, acc[rt], false, false);
      }
    }
    int rb  = (lane >> 4) << 3;
    int col = n0 + (lane & 15);
#pragma unroll
    for (int rt = 0; rt < RTILES; ++rt)
#pragma unroll
      for (int j = 0; j < 8; ++j) {
        float v = acc[rt][j];
        if (relu) v = fmaxf(v, 0.0f);
        int row = (rt << 4) + rb + j;
        if (ybf) ybf[row * ncols + col] = (__bf16)v;
        else     yf [row * ncols + col] = v;
      }
  }
}

// monotone float -> u32 so unsigned-min == float-min; tie -> smaller code wins (matches jnp.argmin)
static __device__ __forceinline__ unsigned fkey(float f) {
  unsigned u = __float_as_uint(f);
  return (u & 0x80000000u) ? ~u : (u | 0x80000000u);
}

// ---------------- prepass kernels ----------------
__global__ void pack_w_bf16T(const float* __restrict__ src, __bf16* __restrict__ dst, int K, int ncols) {
  int e = blockIdx.x * blockDim.x + threadIdx.x;
  if (e >= K * ncols) return;
  int k = e / ncols, c = e - k * ncols;
  dst[(size_t)c * K + k] = (__bf16)src[e];   // coalesced read, transposed write
}

__global__ void pack_cb_bf16(const float* __restrict__ cbs, __bf16* __restrict__ cbbf,
                             float* __restrict__ cb2) {
  int e = blockIdx.x * blockDim.x + threadIdx.x;   // < QN*KCB*L_DIM
  cbbf[e] = (__bf16)cbs[e];
  if ((e & (L_DIM - 1)) == 0) {                    // one lane per code computes ||cb||^2
    const float4* p = (const float4*)(cbs + e);
    float s = 0.0f;
#pragma unroll 8
    for (int i = 0; i < L_DIM / 4; ++i) {
      float4 v = p[i];
      s = fmaf(v.x, v.x, s); s = fmaf(v.y, v.y, s);
      s = fmaf(v.z, v.z, s); s = fmaf(v.w, v.w, s);
    }
    cb2[e >> 7] = s;
  }
}

// ---------------- fused network kernel ----------------
template <bool PACKED>
__global__ void __launch_bounds__(NTHREADS)
imitation_fused(const float* __restrict__ sin, const float* __restrict__ gin,
                const float* pW0, const float* pb0, const float* pW1, const float* pb1,
                const float* pW2, const float* pb2,
                const float* qW0, const float* qb0, const float* qW1, const float* qb1,
                const float* qW2, const float* qb2,
                const float* dW0, const float* db0, const float* dW1, const float* db1,
                const float* dW2, const float* db2,
                const float* __restrict__ cbs,
                const __bf16* __restrict__ wsbf,   // packed weights (unused if !PACKED)
                const float* __restrict__ cb2p,    // ||cb||^2 table (unused if !PACKED)
                float* __restrict__ out)
{
  __shared__ __align__(16) char smem_raw[SMEM_TOTAL];
  __bf16* SBF = (__bf16*)(smem_raw + OFF_SBF);
  __bf16* GBF = (__bf16*)(smem_raw + OFF_GBF);
  __bf16* H0  = (__bf16*)(smem_raw + OFF_H0);
  __bf16* H1  = (__bf16*)(smem_raw + OFF_H1);
  __bf16* HD0 = (__bf16*)(smem_raw + OFF_H0);      // 64x1024, reuses H0+H1
  float*  ZPf = (float*)(smem_raw + OFF_ZP);       // zp, later z_hat accumulator
  float*  Zf  = (float*)(smem_raw + OFF_Z);        // z, later residual
  unsigned long long* MINST = (unsigned long long*)(smem_raw + OFF_MIN);
  __bf16* RBF    = (__bf16*)(smem_raw + OFF_GBF);  // 64x128 bf16 mirror (reuses goal)
  __bf16* CHUNKB = (__bf16*)(smem_raw + OFF_SBF);  // 64x256, reuses SBF after decoder L0
  float*  AACC   = (float*)(smem_raw + OFF_GBF);   // 64x32 f32, reuses GBF

  const int tid  = threadIdx.x;
  const int wave = tid >> 5;
  const int lane = tid & 31;
  const size_t row0g = (size_t)blockIdx.x * MT;

  // ---- stage inputs: fp32 -> bf16 LDS ----
  {
    const float4* sp = (const float4*)(sin + row0g * S_DIM);
    for (int e = tid; e < MT * S_DIM / 4; e += NTHREADS) {
      float4 v = sp[e]; int b = e << 2;
      SBF[b] = (__bf16)v.x; SBF[b + 1] = (__bf16)v.y; SBF[b + 2] = (__bf16)v.z; SBF[b + 3] = (__bf16)v.w;
    }
    const float4* gp = (const float4*)(gin + row0g * G_DIM);
    for (int e = tid; e < MT * G_DIM / 4; e += NTHREADS) {
      float4 v = gp[e]; int b = e << 2;
      GBF[b] = (__bf16)v.x; GBF[b + 1] = (__bf16)v.y; GBF[b + 2] = (__bf16)v.z; GBF[b + 3] = (__bf16)v.w;
    }
  }
  __syncthreads();

#define BP(OFS, LDK, FPTR, FLD) BSrc{ PACKED ? wsbf + (size_t)(OFS) : nullptr, (LDK), (FPTR), (FLD) }

  // ---- prior: s -> H0 -> H1 -> zp ----
  { ASrc a{SBF, S_DIM, 1 << 30, nullptr, 0};
    dense_layer<PACKED>(a, S_DIM, BP(OFS_PW0T, 256, pW0, 512), pb0, H_DIM, H0, nullptr, true, wave, lane); }
  __syncthreads();
  { ASrc a{H0, H_DIM, 1 << 30, nullptr, 0};
    dense_layer<PACKED>(a, H_DIM, BP(OFS_PW1T, 512, pW1, 512), pb1, H_DIM, H1, nullptr, true, wave, lane); }
  __syncthreads();
  { ASrc a{H1, H_DIM, 1 << 30, nullptr, 0};
    dense_layer<PACKED>(a, H_DIM, BP(OFS_PW2T, 512, pW2, 128), pb2, L_DIM, nullptr, ZPf, false, wave, lane); }
  __syncthreads();

  // ---- posterior: [s,goal] -> H0 -> H1 -> z ----
  { ASrc a{SBF, S_DIM, S_DIM, GBF, G_DIM};
    dense_layer<PACKED>(a, S_DIM + G_DIM, BP(OFS_QW0T, 384, qW0, 512), qb0, H_DIM, H0, nullptr, true, wave, lane); }
  __syncthreads();
  { ASrc a{H0, H_DIM, 1 << 30, nullptr, 0};
    dense_layer<PACKED>(a, H_DIM, BP(OFS_QW1T, 512, qW1, 512), qb1, H_DIM, H1, nullptr, true, wave, lane); }
  __syncthreads();
  { ASrc a{H1, H_DIM, 1 << 30, nullptr, 0};
    dense_layer<PACKED>(a, H_DIM, BP(OFS_QW2T, 512, qW2, 128), qb2, L_DIM, nullptr, Zf, false, wave, lane); }
  __syncthreads();

  // ---- residual: Zf := z - zp ; ZPf accumulates z_hat = zp + sum(quant) ----
  for (int e = tid; e < MT * L_DIM; e += NTHREADS) Zf[e] -= ZPf[e];
  __syncthreads();

  // ---- residual VQ, Q quantizers ----
  for (int q = 0; q < QN; ++q) {
    const float*  cbq  = cbs + (size_t)q * KCB * L_DIM;
    const __bf16* cbbq = PACKED ? wsbf + (size_t)OFS_CBBF + (size_t)q * KCB * L_DIM : nullptr;

    for (int e = tid; e < MT; e += NTHREADS) MINST[e] = 0xFFFFFFFFFFFFFFFFull;
    for (int e = tid; e < MT * L_DIM; e += NTHREADS) RBF[e] = (__bf16)Zf[e];  // bf16 mirror of residual
    __syncthreads();

    float bestd[RTILES][8];
    int   besti[RTILES][8];
#pragma unroll
    for (int rt = 0; rt < RTILES; ++rt)
#pragma unroll
      for (int j = 0; j < 8; ++j) { bestd[rt][j] = 3.4e38f; besti[rt][j] = 0; }

    for (int ct = wave; ct < (KCB >> 4); ct += NWAVES) {
      int c0 = ct << 4;
      v8f acc[RTILES];
#pragma unroll
      for (int rt = 0; rt < RTILES; ++rt)
#pragma unroll
        for (int j = 0; j < 8; ++j) acc[rt][j] = 0.0f;

      for (int k0 = 0; k0 < L_DIM; k0 += 32) {
        v16bf b;
        if constexpr (PACKED) b = frag_b_bf(cbbq, L_DIM, k0, c0, lane);
        else                  b = frag_b_cb(cbq, k0, c0, lane);
#pragma unroll
        for (int rt = 0; rt < RTILES; ++rt) {
          v16bf a = frag_a_bf(RBF, L_DIM, rt << 4, k0, lane);
          acc[rt] = __builtin_amdgcn_wmma_f32_16x16x32_bf16(false, a, false, b, (short)0, acc[rt], false, false);
        }
      }
      // ||cb[code]||^2 for this lane's code (precomputed in prepass when packed)
      int code = c0 + (lane & 15);
      float cb2;
      if constexpr (PACKED) {
        cb2 = cb2p[q * KCB + code];
      } else {
        const float4* cp = (const float4*)(cbq + (size_t)code * L_DIM);
        cb2 = 0.0f;
#pragma unroll 8
        for (int i = 0; i < L_DIM / 4; ++i) {
          float4 v = cp[i];
          cb2 = fmaf(v.x, v.x, cb2); cb2 = fmaf(v.y, v.y, cb2);
          cb2 = fmaf(v.z, v.z, cb2); cb2 = fmaf(v.w, v.w, cb2);
        }
      }
#pragma unroll
      for (int rt = 0; rt < RTILES; ++rt)
#pragma unroll
        for (int j = 0; j < 8; ++j) {
          float d = cb2 - 2.0f * acc[rt][j];
          if (d < bestd[rt][j]) { bestd[rt][j] = d; besti[rt][j] = code; }
        }
    }
    // flush per-lane running argmin into per-row packed min (ds_min_u64)
    {
      int rb = (lane >> 4) << 3;
#pragma unroll
      for (int rt = 0; rt < RTILES; ++rt)
#pragma unroll
        for (int j = 0; j < 8; ++j) {
          int row = (rt << 4) + rb + j;
          unsigned long long key =
              ((unsigned long long)fkey(bestd[rt][j]) << 32) | (unsigned)besti[rt][j];
          atomicMin(&MINST[row], key);
        }
    }
    __syncthreads();

    // gather chosen code (fp32), accumulate z_hat, update residual
    for (int e = tid; e < MT * L_DIM; e += NTHREADS) {
      int row = e >> 7;
      int idx = (int)(MINST[row] & 0xFFFFFFFFull);
      float qv = cbq[(size_t)idx * L_DIM + (e & (L_DIM - 1))];
      ZPf[e] += qv;   // z_hat accumulation (quant_st value == quant)
      Zf[e]  -= qv;   // new residual
    }
    __syncthreads();
  }

  // ---- z_hat -> bf16 mirror for decoder input ----
  for (int e = tid; e < MT * L_DIM; e += NTHREADS) RBF[e] = (__bf16)ZPf[e];
  __syncthreads();

  // ---- decoder L0: [s, z_hat] -> HD0 (64x1024) ----
  { ASrc a{SBF, S_DIM, S_DIM, RBF, L_DIM};
    dense_layer<PACKED>(a, S_DIM + L_DIM, BP(OFS_DW0T, 384, dW0, 1024), db0, DH_DIM, HD0, nullptr, true, wave, lane); }
  __syncthreads();

  // init a_hat accumulator with db2 (RBF dead now; AACC overlays same region)
  for (int e = tid; e < MT * A_DIM; e += NTHREADS) AACC[e] = db2[e & (A_DIM - 1)];
  __syncthreads();

  // ---- decoder L1 in 256-col chunks, each immediately folded into final layer ----
  for (int c = 0; c < DH_DIM / CHUNK_N; ++c) {
    { ASrc a{HD0, DH_DIM, 1 << 30, nullptr, 0};
      BSrc b1{ PACKED ? wsbf + (size_t)OFS_DW1T + (size_t)c * CHUNK_N * DH_DIM : nullptr, DH_DIM,
               dW1 + (size_t)c * CHUNK_N, DH_DIM };
      dense_layer<PACKED>(a, DH_DIM, b1, db1 + c * CHUNK_N, CHUNK_N, CHUNKB, nullptr, true, wave, lane); }
    __syncthreads();

    // final layer partial: each wave owns one K=32 slice of this chunk
    {
      int k0 = wave * 32;
      BSrc b2{ PACKED ? wsbf + (size_t)OFS_DW2T + (size_t)c * CHUNK_N : nullptr, DH_DIM,
               dW2 + (size_t)c * CHUNK_N * A_DIM, A_DIM };
#pragma unroll
      for (int ct2 = 0; ct2 < A_DIM / 16; ++ct2) {
        int n0 = ct2 << 4;
        v8f acc[RTILES];
#pragma unroll
        for (int rt = 0; rt < RTILES; ++rt)
#pragma unroll
          for (int j = 0; j < 8; ++j) acc[rt][j] = 0.0f;
        v16bf b = frag_b<PACKED>(b2, k0, n0, lane);
#pragma unroll
        for (int rt = 0; rt < RTILES; ++rt) {
          v16bf a = frag_a_bf(CHUNKB, CHUNK_N, rt << 4, k0, lane);
          acc[rt] = __builtin_amdgcn_wmma_f32_16x16x32_bf16(false, a, false, b, (short)0, acc[rt], false, false);
        }
        int rb = (lane >> 4) << 3, col = n0 + (lane & 15);
#pragma unroll
        for (int rt = 0; rt < RTILES; ++rt)
#pragma unroll
          for (int j = 0; j < 8; ++j)
            atomicAdd(&AACC[((rt << 4) + rb + j) * A_DIM + col], acc[rt][j]);
      }
    }
    __syncthreads();
  }

  // ---- write a_hat ----
  for (int e = tid; e < MT * A_DIM; e += NTHREADS)
    out[row0g * A_DIM + e] = AACC[e];
}

extern "C" void kernel_launch(void* const* d_in, const int* in_sizes, int n_in,
                              void* d_out, int out_size, void* d_ws, size_t ws_size,
                              hipStream_t stream) {
  (void)n_in; (void)out_size;
  const float* s    = (const float*)d_in[0];
  const float* goal = (const float*)d_in[1];
  const float* pW0  = (const float*)d_in[2];  const float* pb0 = (const float*)d_in[3];
  const float* pW1  = (const float*)d_in[4];  const float* pb1 = (const float*)d_in[5];
  const float* pW2  = (const float*)d_in[6];  const float* pb2 = (const float*)d_in[7];
  const float* qW0  = (const float*)d_in[8];  const float* qb0 = (const float*)d_in[9];
  const float* qW1  = (const float*)d_in[10]; const float* qb1 = (const float*)d_in[11];
  const float* qW2  = (const float*)d_in[12]; const float* qb2 = (const float*)d_in[13];
  const float* dW0  = (const float*)d_in[14]; const float* db0 = (const float*)d_in[15];
  const float* dW1  = (const float*)d_in[16]; const float* db1 = (const float*)d_in[17];
  const float* dW2  = (const float*)d_in[18]; const float* db2 = (const float*)d_in[19];
  const float* cbs  = (const float*)d_in[20];
  float* out = (float*)d_out;

  int N = in_sizes[0] / S_DIM;
  dim3 grid(N / MT);

  if (d_ws && ws_size >= WS_NEED) {
    __bf16* wsbf = (__bf16*)d_ws;
    float*  cb2p = (float*)((char*)d_ws + OFS_CB2B);
    auto pack = [&](const float* W, size_t ofs, int K, int C) {
      int tot = K * C;
      pack_w_bf16T<<<(tot + 255) / 256, 256, 0, stream>>>(W, wsbf + ofs, K, C);
    };
    pack(pW0, OFS_PW0T, 256, 512);
    pack(pW1, OFS_PW1T, 512, 512);
    pack(pW2, OFS_PW2T, 512, 128);
    pack(qW0, OFS_QW0T, 384, 512);
    pack(qW1, OFS_QW1T, 512, 512);
    pack(qW2, OFS_QW2T, 512, 128);
    pack(dW0, OFS_DW0T, 384, 1024);
    pack(dW1, OFS_DW1T, 1024, 1024);
    pack(dW2, OFS_DW2T, 1024, 32);
    pack_cb_bf16<<<(QN * KCB * L_DIM) / 256, 256, 0, stream>>>(cbs, wsbf + OFS_CBBF, cb2p);

    imitation_fused<true><<<grid, NTHREADS, 0, stream>>>(
        s, goal, pW0, pb0, pW1, pb1, pW2, pb2,
        qW0, qb0, qW1, qb1, qW2, qb2,
        dW0, db0, dW1, db1, dW2, db2, cbs, wsbf, cb2p, out);
  } else {
    imitation_fused<false><<<grid, NTHREADS, 0, stream>>>(
        s, goal, pW0, pb0, pW1, pb1, pW2, pb2,
        qW0, qb0, qW1, qb1, qW2, qb2,
        dW0, db0, dW1, db1, dW2, db2, cbs, nullptr, nullptr, out);
  }
}
